// EpisodicMemory_71863392797048
// MI455X (gfx1250) — matrix operand order repaired and checked by reading, not compiled, not run on hardware
//
#include <hip/hip_runtime.h>
#include <hip/hip_bf16.h>

typedef __attribute__((ext_vector_type(16))) _Float16 v16h;
typedef __attribute__((ext_vector_type(8)))  _Float16 v8h;
typedef __attribute__((ext_vector_type(8)))  float    v8f;
typedef __attribute__((ext_vector_type(4)))  float    v4f;

#define NHOP 2
#define NMEM 64
#define DIM  32
#define KFEAT 128          // 4*DIM feature vector per memory
#define NTHREADS 256       // 8 waves (wave32)

// Branch-free tanh: native V_TANH_F32 when the builtin exists (CDNA5 has
// TANH in the TRANS32 op list), else exp-based straight-line fallback.
__device__ __forceinline__ float fast_tanh(float x) {
#if __has_builtin(__builtin_amdgcn_tanhf)
    return __builtin_amdgcn_tanhf(x);
#else
    const float ax = fabsf(x);
    const float e  = __expf(-2.0f * ax);
    const float t  = 1.0f - 2.0f * e * __builtin_amdgcn_rcpf(1.0f + e);
    return copysignf(t, x);
#endif
}

__device__ __forceinline__ float fast_sigmoid(float x) {
    return __builtin_amdgcn_rcpf(1.0f + __expf(-x));
}

__global__ __launch_bounds__(NTHREADS)
void episodic_memory_kernel(const float* __restrict__ hs,   // (B,2,64,32)
                            const float* __restrict__ Rs,   // (B,2,64,32,32)
                            const float* __restrict__ ts,   // (B,2,64,32)
                            const float* __restrict__ vs,   // (B,32)
                            const float* __restrict__ W1,   // (32,128)
                            const float* __restrict__ b1,   // (32)
                            const float* __restrict__ W2,   // (1,32)
                            const float* __restrict__ b2,   // (1)
                            const float* __restrict__ Wih,  // (2,96,32)
                            const float* __restrict__ Whh,  // (2,96,32)
                            const float* __restrict__ bih,  // (2,96)
                            const float* __restrict__ bhh,  // (2,96)
                            float* __restrict__ out)        // (B,32)
{
    __shared__ float     h_s[NMEM * DIM];         // 8 KB
    __shared__ float     Rt_s[8][DIM * DIM];      // 32 KB  per-wave R tile
    __shared__ float     Rh_s[NMEM * DIM];        // 8 KB
    __shared__ _Float16  z_s[NMEM * KFEAT];       // 16 KB  f16 features (WMMA A)
    __shared__ _Float16  W1h_s[DIM * KFEAT];      // 8 KB   f16 weights (WMMA B)
    __shared__ float     P_s[NMEM * DIM];         // 8 KB   tanh(.)*W2 partials
    __shared__ float     Z_s[NMEM];
    __shared__ float     g_s[NMEM];
    __shared__ float     o_s[DIM];
    __shared__ float     M_s[DIM];
    __shared__ float     v_s[DIM];
    __shared__ float     b1_s[DIM];
    __shared__ float     W2_s[DIM];
    __shared__ float     gi_s[3 * DIM];
    __shared__ float     gh_s[3 * DIM];

    const int b    = blockIdx.x;
    const int tid  = threadIdx.x;
    const int wave = tid >> 5;
    const int lane = tid & 31;

    // ---- per-batch constants ------------------------------------------------
    if (tid < DIM) {
        float vv  = vs[b * DIM + tid];
        v_s[tid]  = vv;
        M_s[tid]  = vv;          // M starts as vs
        b1_s[tid] = b1[tid];
        W2_s[tid] = W2[tid];
    }
    for (int i = tid; i < DIM * KFEAT; i += NTHREADS)
        W1h_s[i] = (_Float16)W1[i];
    const float b2v = b2[0];
    __syncthreads();

    for (int hop = 0; hop < NHOP; ++hop) {
        const size_t hopbase = ((size_t)b * NHOP + hop) * (size_t)(NMEM * DIM);

        // stage h tile for this hop (coalesced)
        for (int i = tid; i < NMEM * DIM; i += NTHREADS)
            h_s[i] = hs[hopbase + i];
        __syncthreads();

        // ---- Phase A: Rh[m][d] = sum_e R[m][d][e] * h[m][e] -----------------
        // Rs is streamed exactly once -> non-temporal b128 loads.
        const float* Rbase = Rs + ((size_t)b * NHOP + hop) * (size_t)(NMEM * DIM * DIM);
        float* myRt = Rt_s[wave];
        for (int i = 0; i < 8; ++i) {
            const int m = wave * 8 + i;
            const float* rsrc = Rbase + (size_t)m * (DIM * DIM);
            if (i < 7) __builtin_prefetch(rsrc + DIM * DIM, 0, 0); // next tile
            #pragma unroll
            for (int c = 0; c < 8; ++c) {
                const int linear = c * 128 + lane * 4;           // (d,e) flat
                v4f val = __builtin_nontemporal_load(
                              reinterpret_cast<const v4f*>(rsrc + linear));
                *reinterpret_cast<v4f*>(myRt + linear) = val;
            }
            // lane == d ; rotate e by lane -> conflict-free LDS reads
            float acc = 0.0f;
            #pragma unroll
            for (int k = 0; k < DIM; ++k) {
                const int e = (k + lane) & (DIM - 1);
                acc += myRt[lane * DIM + e] * h_s[m * DIM + e];
            }
            Rh_s[m * DIM + lane] = acc;
        }
        __syncthreads();

        // ---- Phase B1: build z features in f16 ------------------------------
        for (int idx = tid; idx < NMEM * KFEAT; idx += NTHREADS) {
            const int m   = idx >> 7;            // /128
            const int k   = idx & (KFEAT - 1);
            const int d0  = k & (DIM - 1);
            const int sel = k >> 5;
            const float rh = Rh_s[m * DIM + d0];
            const float vv = v_s[d0];
            const float pm = M_s[d0];
            float val;
            if      (sel == 0) val = rh * vv;
            else if (sel == 1) val = rh * pm;
            else if (sel == 2) val = fabsf(rh - vv);
            else               val = fabsf(rh - pm);
            z_s[idx] = (_Float16)val;
        }
        __syncthreads();

        // ---- Phase B2: S = z @ W1.T via v_wmma_f32_16x16x32_f16 -------------
        // one 16x16 output tile per wave: mt in [0,4), nt in [0,2)
        {
            const int mt    = wave >> 1;
            const int nt    = wave & 1;
            const int hi    = (lane < 16) ? 0 : 1;
            const int arow  = mt * 16 + (lane & 15);   // A row (M) per ISA layout
            const int abase = hi * 8;                  // A K sub-offset
            const int bcol  = nt * 16 + (lane & 15);   // B column (N)
            const int bbase = hi * 8;                  // B K sub-offset
            v8f acc = {};
            #pragma unroll
            for (int ki = 0; ki < 4; ++ki) {
                const int k0 = ki * 32;
                const v8h a_lo = *reinterpret_cast<const v8h*>(&z_s[arow * KFEAT + k0 + abase]);
                const v8h a_hi = *reinterpret_cast<const v8h*>(&z_s[arow * KFEAT + k0 + abase + 16]);
                const v8h b_lo = *reinterpret_cast<const v8h*>(&W1h_s[bcol * KFEAT + k0 + bbase]);
                const v8h b_hi = *reinterpret_cast<const v8h*>(&W1h_s[bcol * KFEAT + k0 + bbase + 16]);
                v16h af = __builtin_shufflevector(a_lo, a_hi,
                          0,1,2,3,4,5,6,7,8,9,10,11,12,13,14,15);
                v16h bf = __builtin_shufflevector(b_lo, b_hi,
                          0,1,2,3,4,5,6,7,8,9,10,11,12,13,14,15);
                acc = __builtin_amdgcn_wmma_f32_16x16x32_f16(
                          false, af, false, bf, (short)0, acc, false, false);
            }
            // epilogue: tanh(S + b1) * W2 -> partial-product matrix P
            const int mrow0 = mt * 16 + ((lane < 16) ? 0 : 8);
            const int ncol  = nt * 16 + (lane & 15);
            #pragma unroll
            for (int r = 0; r < 8; ++r) {
                const float t = fast_tanh(acc[r] + b1_s[ncol]);
                P_s[(mrow0 + r) * DIM + ncol] = t * W2_s[ncol];
            }
        }
        __syncthreads();

        // scores Z[m] = sum_d P[m][d] + b2  (rotated index -> no bank conflicts)
        if (tid < NMEM) {
            float z = b2v;
            #pragma unroll
            for (int k = 0; k < DIM; ++k) {
                const int d = (k + tid) & (DIM - 1);
                z += P_s[tid * DIM + d];
            }
            Z_s[tid] = z;
        }
        __syncthreads();

        // wave-parallel softmax over the 64 memories (wave 0; 2 values/lane)
        if (wave == 0) {
            const float z0 = Z_s[lane];
            const float z1 = Z_s[lane + 32];
            float mx = fmaxf(z0, z1);
            #pragma unroll
            for (int off = 16; off > 0; off >>= 1)
                mx = fmaxf(mx, __shfl_xor(mx, off, 32));
            const float e0 = __expf(z0 - mx);
            const float e1 = __expf(z1 - mx);
            float s = e0 + e1;
            #pragma unroll
            for (int off = 16; off > 0; off >>= 1)
                s += __shfl_xor(s, off, 32);
            const float inv = __builtin_amdgcn_rcpf(s);
            g_s[lane]      = e0 * inv;
            g_s[lane + 32] = e1 * inv;
        }
        __syncthreads();

        // ---- Phase D: o[d] = sum_m t[m][d] * g[m] ---------------------------
        if (tid < DIM) {
            const float* tsrc = ts + hopbase;
            float o = 0.0f;
            for (int m = 0; m < NMEM; ++m)
                o += tsrc[m * DIM + tid] * g_s[m];
            o_s[tid] = o;
        }
        __syncthreads();

        // ---- Phase E: GRU cell M = GRU(o, M) --------------------------------
        if (tid < 3 * DIM) {
            const float* wi = Wih + ((size_t)hop * 3 * DIM + tid) * DIM;
            const float* wh = Whh + ((size_t)hop * 3 * DIM + tid) * DIM;
            float gi = bih[hop * 3 * DIM + tid];
            float gh = bhh[hop * 3 * DIM + tid];
            #pragma unroll
            for (int e = 0; e < DIM; ++e) {
                gi += o_s[e] * wi[e];
                gh += M_s[e] * wh[e];
            }
            gi_s[tid] = gi;
            gh_s[tid] = gh;
        }
        __syncthreads();
        if (tid < DIM) {
            const float r  = fast_sigmoid(gi_s[tid] + gh_s[tid]);
            const float zg = fast_sigmoid(gi_s[DIM + tid] + gh_s[DIM + tid]);
            const float n  = fast_tanh(gi_s[2 * DIM + tid] + r * gh_s[2 * DIM + tid]);
            M_s[tid] = (1.0f - zg) * n + zg * M_s[tid];
        }
        __syncthreads();
    }

    if (tid < DIM) out[b * DIM + tid] = M_s[tid];
}

extern "C" void kernel_launch(void* const* d_in, const int* in_sizes, int n_in,
                              void* d_out, int out_size, void* d_ws, size_t ws_size,
                              hipStream_t stream) {
    (void)n_in; (void)out_size; (void)d_ws; (void)ws_size;
    const int B = in_sizes[3] / DIM;   // vs is (B, DIM)
    episodic_memory_kernel<<<dim3(B), dim3(NTHREADS), 0, stream>>>(
        (const float*)d_in[0],  // hs
        (const float*)d_in[1],  // Rs
        (const float*)d_in[2],  // ts
        (const float*)d_in[3],  // vs
        (const float*)d_in[4],  // W1
        (const float*)d_in[5],  // b1
        (const float*)d_in[6],  // W2
        (const float*)d_in[7],  // b2
        (const float*)d_in[8],  // W_ih
        (const float*)d_in[9],  // W_hh
        (const float*)d_in[10], // b_ih
        (const float*)d_in[11], // b_hh
        (float*)d_out);
}
